// WindAdvectionBlock_15324443312137
// MI455X (gfx1250) — matrix operand order repaired and checked by reading, not compiled, not run on hardware
//
#include <hip/hip_runtime.h>

// ---------------------------------------------------------------------------
// Problem constants (from reference setup_inputs)
// ---------------------------------------------------------------------------
#define BB   8
#define CC   256
#define MID  64
#define HH   128
#define WW   128
#define HWSZ 16384          // H*W
#define KTOT 576            // MID*9  (deform GEMM K)

typedef __attribute__((ext_vector_type(16))) __bf16 v16bf;
typedef __attribute__((ext_vector_type(8)))  float  v8f;

union BF16x16 {                 // one WMMA A/B fragment per lane (16 bf16)
    v16bf          v;
    unsigned short h[16];
    uint4          q[2];
};
union Pack8 { unsigned short h[8]; uint4 q; };

__device__ __forceinline__ unsigned short f2bf(float f) {
    unsigned u = __float_as_uint(f);
    u += 0x7fffu + ((u >> 16) & 1u);          // round-to-nearest-even
    return (unsigned short)(u >> 16);
}
__device__ __forceinline__ float bf2f(unsigned short h) {
    return __uint_as_float(((unsigned)h) << 16);
}
__device__ __forceinline__ float gelu_exact(float x) {
    return 0.5f * x * (1.0f + erff(x * 0.70710678118654752440f));
}
// A-fragment 16x32 layout (contiguous K, matches B): lane 0-15 -> K {0..15}; lane 16-31 -> K {16..31}
__device__ __forceinline__ int a_klocal(int e, int lh) {
    return e + 16 * lh;
}
__device__ __forceinline__ v8f wmma_bf16(const BF16x16& a, const BF16x16& b, v8f c) {
    return __builtin_amdgcn_wmma_f32_16x16x32_bf16(false, a.v, false, b.v,
                                                   (short)0, c, false, false);
}
__device__ __forceinline__ void load_frag(const unsigned short* p, BF16x16& f) {
    const uint4* q = (const uint4*)p;
    f.q[0] = q[0];
    f.q[1] = q[1];
}

// ---------------------------------------------------------------------------
// k_prep: swizzle the three GEMM weight matrices into A-fragment layout
//   wc: [4 mt][8 t][32 lane][16 e]   from compress_w [64][256]
//   wd: [4 mt][18 t][32][16]         from deform_w   [64][576] (c*9+k flat)
//   we: [16 mt][2 t][32][16]         from expand_w   [256][64]
// also zeroes group-norm partial/stat scratch
// ---------------------------------------------------------------------------
__global__ void k_prep(const float* __restrict__ cw, const float* __restrict__ dw,
                       const float* __restrict__ ew,
                       unsigned short* __restrict__ wc, unsigned short* __restrict__ wd,
                       unsigned short* __restrict__ we,
                       float* __restrict__ partials, float* __restrict__ stats) {
    int idx = blockIdx.x * blockDim.x + threadIdx.x;
    if (idx < 1024) partials[idx] = 0.0f;
    if (idx < 64)   stats[idx]    = 0.0f;

    if (idx < 16384) {                                   // compress weights
        int e = idx & 15, l = (idx >> 4) & 31, rest = idx >> 9;
        int t = rest & 7, mt = rest >> 3;
        int kl = a_klocal(e, l >> 4);
        wc[idx] = f2bf(cw[(size_t)(mt * 16 + (l & 15)) * 256 + t * 32 + kl]);
    } else if (idx < 16384 + 36864) {                    // deform weights
        int j = idx - 16384;
        int e = j & 15, l = (j >> 4) & 31, rest = j >> 9;
        int t = rest % 18, mt = rest / 18;
        int kl = a_klocal(e, l >> 4);
        wd[j] = f2bf(dw[(size_t)(mt * 16 + (l & 15)) * KTOT + t * 32 + kl]);
    } else if (idx < 16384 + 36864 + 32768) {            // expand weights
        int j = idx - 16384 - 36864;
        int e = j & 15, l = (j >> 4) & 31, rest = j >> 9;
        int t = rest & 1, mt = rest >> 1;
        int kl = a_klocal(e, l >> 4);
        we[j] = f2bf(ew[(size_t)(mt * 16 + (l & 15)) * 64 + t * 32 + kl]);
    }
}

// ---------------------------------------------------------------------------
// k_off1: conv3x3 (2 -> 32 ch), pad 1, exact GELU.  h layout [b][32][HW]
// ---------------------------------------------------------------------------
__global__ void k_off1(const float* __restrict__ wind, const float* __restrict__ w,
                       const float* __restrict__ bias, float* __restrict__ hout) {
    int idx = blockIdx.x * blockDim.x + threadIdx.x;
    if (idx >= BB * 32 * HWSZ) return;
    int p  = idx & (HWSZ - 1);
    int co = (idx >> 14) & 31;
    int b  = idx >> 19;
    int y = p >> 7, x = p & 127;
    const float* src = wind + (size_t)b * 2 * HWSZ;
    float acc = bias[co];
    for (int ci = 0; ci < 2; ci++)
        for (int dy = -1; dy <= 1; dy++) {
            int yy = y + dy; if (yy < 0 || yy >= HH) continue;
            for (int dx = -1; dx <= 1; dx++) {
                int xx = x + dx; if (xx < 0 || xx >= WW) continue;
                acc += src[(size_t)ci * HWSZ + yy * WW + xx] *
                       w[(co * 2 + ci) * 9 + (dy + 1) * 3 + (dx + 1)];
            }
        }
    hout[idx] = gelu_exact(acc);
}

// ---------------------------------------------------------------------------
// k_off2: conv3x3 (32 -> 18 ch), pad 1.  offsets layout [b][18][HW]
// ---------------------------------------------------------------------------
__global__ void k_off2(const float* __restrict__ hin, const float* __restrict__ w,
                       const float* __restrict__ bias, float* __restrict__ offs) {
    int idx = blockIdx.x * blockDim.x + threadIdx.x;
    if (idx >= BB * 18 * HWSZ) return;
    int p  = idx & (HWSZ - 1);
    int co = (idx >> 14) % 18;
    int b  = idx / (18 * HWSZ);
    int y = p >> 7, x = p & 127;
    const float* src = hin + (size_t)b * 32 * HWSZ;
    float acc = bias[co];
    for (int ci = 0; ci < 32; ci++) {
        const float* s = src + (size_t)ci * HWSZ;
        const float* wk = w + (co * 32 + ci) * 9;
        for (int dy = -1; dy <= 1; dy++) {
            int yy = y + dy; if (yy < 0 || yy >= HH) continue;
            for (int dx = -1; dx <= 1; dx++) {
                int xx = x + dx; if (xx < 0 || xx >= WW) continue;
                acc += s[yy * WW + xx] * wk[(dy + 1) * 3 + (dx + 1)];
            }
        }
    }
    offs[idx] = acc;
}

// ---------------------------------------------------------------------------
// k_compress: 1x1 conv as GEMM  xc[m=64][p] = Wc[64][256] * X[256][p]
// one wave computes M=64 x N=16 tile; software pipeline: A-frags batched,
// next B column issued before the 4 back-to-back WMMAs.
// output stored PIXEL-MAJOR bf16: xc[b][p][c] (64ch = one 128B line).
// ---------------------------------------------------------------------------
__global__ void __launch_bounds__(256) k_compress(const float* __restrict__ feat,
                                                  const unsigned short* __restrict__ wcs,
                                                  const float* __restrict__ cb,
                                                  unsigned short* __restrict__ xc) {
    int gtid = blockIdx.x * blockDim.x + threadIdx.x;
    int lane = gtid & 31, wave = gtid >> 5;
    int b  = wave / (HWSZ / 16);
    int p0 = (wave % (HWSZ / 16)) * 16;
    int lm = lane & 15, lh = lane >> 4;
    const float* colbase = feat + (size_t)b * CC * HWSZ + (size_t)(lh * 16) * HWSZ + p0 + lm;

    v8f acc[4];
    for (int mt = 0; mt < 4; mt++)
        for (int j = 0; j < 8; j++) acc[mt][j] = 0.0f;

    float braw[16];
#pragma unroll
    for (int e = 0; e < 16; e++) braw[e] = colbase[(size_t)e * HWSZ];

    for (int t = 0; t < 8; t++) {
        BF16x16 afr[4];
#pragma unroll
        for (int mt = 0; mt < 4; mt++)
            load_frag(wcs + (size_t)((mt * 8 + t) * 32 + lane) * 16, afr[mt]);

        BF16x16 bfr;                                   // B frag: element e <-> K = e + 16*lh
#pragma unroll
        for (int e = 0; e < 16; e++) bfr.h[e] = f2bf(braw[e]);

        if (t < 7) {                                   // prefetch next B column
            const float* col = colbase + (size_t)(t + 1) * 32 * HWSZ;
#pragma unroll
            for (int e = 0; e < 16; e++) braw[e] = col[(size_t)e * HWSZ];
        }
#pragma unroll
        for (int mt = 0; mt < 4; mt++) acc[mt] = wmma_bf16(afr[mt], bfr, acc[mt]);
    }

    unsigned short* xcb = xc + (size_t)b * HWSZ * 64;
    for (int mt = 0; mt < 4; mt++) {
        Pack8 pk;
        for (int r = 0; r < 8; r++) {                  // D: VGPR r <-> m = r + 8*lh
            int c = mt * 16 + lh * 8 + r;
            pk.h[r] = f2bf(acc[mt][r] + cb[c]);
        }
        *(uint4*)(xcb + (size_t)(p0 + lm) * 64 + mt * 16 + lh * 8) = pk.q;
    }
}

// ---------------------------------------------------------------------------
// k_deform: bilinear sample 9 taps into LDS (B-fragment layout), then WMMA
// GEMM with K=576 (LDS B double-buffered).  One wave per 16-pixel strip;
// 2 waves / 64-thread block.  y output [b][64][HW] bf16.
// ---------------------------------------------------------------------------
__global__ void __launch_bounds__(64) k_deform(const unsigned short* __restrict__ xc,
                                               const float* __restrict__ offs,
                                               const unsigned short* __restrict__ wds,
                                               const float* __restrict__ db,
                                               unsigned short* __restrict__ ybuf) {
    __shared__ __align__(16) unsigned short lds[2 * 18 * 32 * 16];   // 36 KB
    int tid  = threadIdx.x;
    int lane = tid & 31, wv = tid >> 5;
    unsigned short* sm = lds + (size_t)wv * 18 * 32 * 16;

    int wave = blockIdx.x * 2 + wv;
    int b    = wave / (HWSZ / 16);
    int p0   = (wave % (HWSZ / 16)) * 16;
    int row  = p0 >> 7;
    int col0 = p0 & 127;
    const float*          ob  = offs + (size_t)b * 18 * HWSZ;
    const unsigned short* xcb = xc   + (size_t)b * HWSZ * 64;

    // ---- gather stage: channels across lanes (2 per lane), (tap,pixel) in loop
    for (int kn = 0; kn < 144; kn++) {
        int k = kn >> 4, n = kn & 15;
        int ki = k / 3 - 1, kj = k % 3 - 1;
        int p = p0 + n;
        float dy = ob[(size_t)(2 * k)     * HWSZ + p];     // uniform across lanes
        float dx = ob[(size_t)(2 * k + 1) * HWSZ + p];
        float py = (float)(row + ki) + dy;
        float px = (float)(col0 + n + kj) + dx;
        float fy = floorf(py), fx = floorf(px);
        float wy = py - fy, wx = px - fx;
        int   y0 = (int)fy,  x0 = (int)fx;
        float v0 = 0.0f, v1 = 0.0f;                        // channels 2*lane, 2*lane+1
        for (int cy = 0; cy < 2; cy++) {
            int yy = y0 + cy; if (yy < 0 || yy >= HH) continue;
            float wyv = cy ? wy : 1.0f - wy;
            for (int cx = 0; cx < 2; cx++) {
                int xx = x0 + cx; if (xx < 0 || xx >= WW) continue;
                float wgt = wyv * (cx ? wx : 1.0f - wx);
                unsigned u = *(const unsigned*)(xcb + (size_t)(yy * WW + xx) * 64 + lane * 2);
                v0 += wgt * bf2f((unsigned short)(u & 0xffffu));
                v1 += wgt * bf2f((unsigned short)(u >> 16));
            }
        }
        // store to LDS directly in B-fragment layout: frag elem e = K&15,
        // frag lane = n + 16*((K>>4)&1), tile t = K>>5;  K = c*9 + k
        int kk = (2 * lane) * 9 + k;
        {
            int t = kk >> 5, kr = kk & 31;
            sm[(size_t)((t * 32 + ((kr >> 4) << 4) + n)) * 16 + (kr & 15)] = f2bf(v0);
        }
        kk += 9;
        {
            int t = kk >> 5, kr = kk & 31;
            sm[(size_t)((t * 32 + ((kr >> 4) << 4) + n)) * 16 + (kr & 15)] = f2bf(v1);
        }
    }
    __syncthreads();

    // ---- GEMM stage: M=64 (4 tiles) x N=16, K=576 (18 steps), B double-buffered
    int lm = lane & 15, lh = lane >> 4;
    v8f acc[4];
    for (int mt = 0; mt < 4; mt++)
        for (int j = 0; j < 8; j++) acc[mt][j] = 0.0f;

    BF16x16 bcur;
    load_frag(sm + (size_t)lane * 16, bcur);               // t = 0
    for (int t = 0; t < 18; t++) {
        BF16x16 afr[4];
#pragma unroll
        for (int mt = 0; mt < 4; mt++)
            load_frag(wds + (size_t)((mt * 18 + t) * 32 + lane) * 16, afr[mt]);
        BF16x16 bnext;
        if (t < 17)
            load_frag(sm + (size_t)(((t + 1) * 32 + lane)) * 16, bnext);
#pragma unroll
        for (int mt = 0; mt < 4; mt++) acc[mt] = wmma_bf16(afr[mt], bcur, acc[mt]);
        if (t < 17) bcur = bnext;
    }

    unsigned short* yb = ybuf + (size_t)b * 64 * HWSZ;
    for (int mt = 0; mt < 4; mt++)
        for (int r = 0; r < 8; r++) {
            int m = mt * 16 + lh * 8 + r;
            yb[(size_t)m * HWSZ + p0 + lm] = f2bf(acc[mt][r] + db[m]);
        }
}

// ---------------------------------------------------------------------------
// k_expand: 1x1 conv as GEMM  xe[256][p] = We[256][64] * y[64][p]  (+bias)
// 16 M-tiles processed in groups of 4 with preloaded A-fragments.
// ---------------------------------------------------------------------------
__global__ void __launch_bounds__(256) k_expand(const unsigned short* __restrict__ ybuf,
                                                const unsigned short* __restrict__ wes,
                                                const float* __restrict__ eb,
                                                float* __restrict__ xe) {
    int gtid = blockIdx.x * blockDim.x + threadIdx.x;
    int lane = gtid & 31, wave = gtid >> 5;
    int b  = wave / (HWSZ / 16);
    int p0 = (wave % (HWSZ / 16)) * 16;
    int lm = lane & 15, lh = lane >> 4;
    const unsigned short* yb = ybuf + (size_t)b * 64 * HWSZ;

    v8f acc[16];
    for (int mt = 0; mt < 16; mt++)
        for (int j = 0; j < 8; j++) acc[mt][j] = 0.0f;

#pragma unroll
    for (int t = 0; t < 2; t++) {
        BF16x16 bfr;
        const unsigned short* col = yb + (size_t)(t * 32 + lh * 16) * HWSZ + p0 + lm;
#pragma unroll
        for (int e = 0; e < 16; e++) bfr.h[e] = col[(size_t)e * HWSZ];
        for (int mg = 0; mg < 4; mg++) {
            BF16x16 afr[4];
#pragma unroll
            for (int j = 0; j < 4; j++)
                load_frag(wes + (size_t)(((mg * 4 + j) * 2 + t) * 32 + lane) * 16, afr[j]);
#pragma unroll
            for (int j = 0; j < 4; j++)
                acc[mg * 4 + j] = wmma_bf16(afr[j], bfr, acc[mg * 4 + j]);
        }
    }

    float* xb = xe + (size_t)b * CC * HWSZ;
    for (int mt = 0; mt < 16; mt++)
        for (int r = 0; r < 8; r++) {
            int m = mt * 16 + lh * 8 + r;
            xb[(size_t)m * HWSZ + p0 + lm] = acc[mt][r] + eb[m];
        }
}

// ---------------------------------------------------------------------------
// Group-norm reduction (deterministic two-stage, fixed tree order)
// ---------------------------------------------------------------------------
__global__ void __launch_bounds__(256) k_gn_part(const float* __restrict__ xe,
                                                 float* __restrict__ partials) {
    __shared__ float ss[256], ss2[256];
    int wg = blockIdx.x;                 // 0..511 : (b*4+g)*16 + chunk
    int bg = wg >> 4, chunk = wg & 15;
    size_t base = (size_t)bg * 64 * HWSZ + (size_t)chunk * 65536;
    float s = 0.0f, s2 = 0.0f;
    for (int i = threadIdx.x; i < 65536; i += 256) {
        float v = xe[base + i];
        s += v; s2 += v * v;
    }
    ss[threadIdx.x] = s; ss2[threadIdx.x] = s2;
    __syncthreads();
    for (int st = 128; st > 0; st >>= 1) {
        if (threadIdx.x < st) {
            ss[threadIdx.x]  += ss[threadIdx.x + st];
            ss2[threadIdx.x] += ss2[threadIdx.x + st];
        }
        __syncthreads();
    }
    if (threadIdx.x == 0) { partials[wg * 2] = ss[0]; partials[wg * 2 + 1] = ss2[0]; }
}

__global__ void k_gn_final(const float* __restrict__ partials, float* __restrict__ stats) {
    int bg = threadIdx.x;
    if (bg >= 32) return;
    float s = 0.0f, s2 = 0.0f;
    for (int i = 0; i < 16; i++) {
        s  += partials[(bg * 16 + i) * 2];
        s2 += partials[(bg * 16 + i) * 2 + 1];
    }
    const float inv = 1.0f / (64.0f * (float)HWSZ);
    float mean = s * inv;
    float var  = fmaxf(s2 * inv - mean * mean, 0.0f);
    stats[bg * 2]     = mean;
    stats[bg * 2 + 1] = rsqrtf(var + 1e-5f);
}

// ---------------------------------------------------------------------------
// k_out: out = features + gelu( (xe-mean)*rstd*gamma + beta ), float4-wide
// ---------------------------------------------------------------------------
__global__ void __launch_bounds__(256) k_out(const float* __restrict__ feat,
                                             const float* __restrict__ xe,
                                             const float* __restrict__ stats,
                                             const float* __restrict__ gamma,
                                             const float* __restrict__ beta,
                                             float* __restrict__ out) {
    size_t i4 = (size_t)blockIdx.x * blockDim.x + threadIdx.x;
    if (i4 >= (size_t)BB * CC * HWSZ / 4) return;
    size_t i = i4 * 4;
    int c  = (int)((i >> 14) & 255);
    int b  = (int)(i >> 22);
    int bg = b * 4 + (c >> 6);
    float mean = stats[bg * 2], rstd = stats[bg * 2 + 1];
    float ga = gamma[c] * rstd;
    float be = beta[c] - mean * ga;
    float4 xv = *(const float4*)(xe + i);
    float4 fv = *(const float4*)(feat + i);
    float4 o;
    o.x = fv.x + gelu_exact(xv.x * ga + be);
    o.y = fv.y + gelu_exact(xv.y * ga + be);
    o.z = fv.z + gelu_exact(xv.z * ga + be);
    o.w = fv.w + gelu_exact(xv.w * ga + be);
    *(float4*)(out + i) = o;
}

// ---------------------------------------------------------------------------
extern "C" void kernel_launch(void* const* d_in, const int* in_sizes, int n_in,
                              void* d_out, int out_size, void* d_ws, size_t ws_size,
                              hipStream_t stream) {
    const float* features = (const float*)d_in[0];
    const float* wind_uv  = (const float*)d_in[1];
    const float* cw       = (const float*)d_in[2];
    const float* cb       = (const float*)d_in[3];
    const float* o1w      = (const float*)d_in[4];
    const float* o1b      = (const float*)d_in[5];
    const float* o2w      = (const float*)d_in[6];
    const float* o2b      = (const float*)d_in[7];
    const float* dw       = (const float*)d_in[8];
    const float* db       = (const float*)d_in[9];
    const float* ew       = (const float*)d_in[10];
    const float* eb       = (const float*)d_in[11];
    const float* gamma    = (const float*)d_in[12];
    const float* beta     = (const float*)d_in[13];
    float* out            = (float*)d_out;

    char*  ws = (char*)d_ws;
    size_t o  = 0;
    auto take = [&](size_t bytes) -> char* {
        char* p = ws + o;
        o = (o + bytes + 255) & ~(size_t)255;
        return p;
    };
    unsigned short* wc_swz = (unsigned short*)take(16384 * 2);                    // compress A-frags
    unsigned short* wd_swz = (unsigned short*)take(36864 * 2);                    // deform A-frags
    unsigned short* we_swz = (unsigned short*)take(32768 * 2);                    // expand A-frags
    unsigned short* xc     = (unsigned short*)take((size_t)BB * HWSZ * 64 * 2);   // compressed, [b][p][c] bf16
    float*          hbuf   = (float*)take((size_t)BB * 32 * HWSZ * 4);            // offset hidden
    float*          offs   = (float*)take((size_t)BB * 18 * HWSZ * 4);            // offsets
    unsigned short* ybuf   = (unsigned short*)take((size_t)BB * 64 * HWSZ * 2);   // deform out bf16
    float*          xe     = (float*)take((size_t)BB * CC * HWSZ * 4);            // expanded f32
    float*          parts  = (float*)take(1024 * 4);
    float*          stats  = (float*)take(64 * 4);
    (void)in_sizes; (void)n_in; (void)out_size; (void)ws_size;

    k_prep<<<336, 256, 0, stream>>>(cw, dw, ew, wc_swz, wd_swz, we_swz, parts, stats);
    k_off1<<<(BB * 32 * HWSZ) / 256, 256, 0, stream>>>(wind_uv, o1w, o1b, hbuf);
    k_off2<<<(BB * 18 * HWSZ) / 256, 256, 0, stream>>>(hbuf, o2w, o2b, offs);
    k_compress<<<(BB * HWSZ / 16) * 32 / 256, 256, 0, stream>>>(features, wc_swz, cb, xc);
    k_deform<<<(BB * HWSZ / 16) / 2, 64, 0, stream>>>(xc, offs, wd_swz, db, ybuf);
    k_expand<<<(BB * HWSZ / 16) * 32 / 256, 256, 0, stream>>>(ybuf, we_swz, eb, xe);
    k_gn_part<<<512, 256, 0, stream>>>(xe, parts);
    k_gn_final<<<1, 32, 0, stream>>>(parts, stats);
    k_out<<<(BB * CC * HWSZ / 4) / 256, 256, 0, stream>>>(features, xe, stats, gamma, beta, out);
}
